// GroupedSelfAttention_13984413516080
// MI455X (gfx1250) — matrix operand order repaired and checked by reading, not compiled, not run on hardware
//
#include <hip/hip_runtime.h>
#include <hip/hip_bf16.h>

// ---------------------------------------------------------------------------
// GQA self-attention block for MI455X (gfx1250, wave32, WMMA).
// All matmuls: bf16 operands, fp32 accumulation via v_wmma_f32_16x16x32_bf16.
// GEMMs are register-staged double-buffered; flash-attention K/V staging uses
// GLOBAL_LOAD_ASYNC_TO_LDS_B128 when the toolchain exposes the builtin.
// ---------------------------------------------------------------------------

typedef __attribute__((ext_vector_type(16))) __bf16 v16bf;
typedef __attribute__((ext_vector_type(8)))  __bf16 v8bf;
typedef __attribute__((ext_vector_type(8)))  float  v8f;

#define BB   2
#define TT   2048
#define NH   16
#define NKV  4
#define HD   128
#define NE   2048
#define NREP (NH / NKV)

#if defined(__has_builtin)
#if __has_builtin(__builtin_amdgcn_global_load_async_to_lds_b128) && \
    __has_builtin(__builtin_amdgcn_s_wait_asynccnt)
#define USE_ASYNC_LDS 1
#endif
#endif

#ifdef USE_ASYNC_LDS
// Builtin signature (from hipcc diagnostic): param0 is AS1 int4*, LDS dest AS3.
typedef __attribute__((__vector_size__(4 * sizeof(int)))) int i32x4;
typedef __attribute__((address_space(1))) i32x4 as1_i32x4;
typedef __attribute__((address_space(3))) i32x4 as3_i32x4;
#endif

__device__ inline v8f zero8() {
    v8f z;
#pragma unroll
    for (int i = 0; i < 8; ++i) z[i] = 0.0f;
    return z;
}

// A-matrix fragment (16x32 bf16): lane holds row (lane%16);
// VGPR0-3 = K[8*half .. 8*half+7], VGPR4-7 = K[16+8*half .. 16+8*half+7].
__device__ inline v16bf frag_a(const __bf16* base, int half) {
    union { v16bf v; v8bf h[2]; } u;
    u.h[0] = *(const v8bf*)(base + 8 * half);
    u.h[1] = *(const v8bf*)(base + 16 + 8 * half);
    return u.v;
}

// B-matrix fragment (32x16 bf16): lane holds column (lane%16);
// VGPR0-3 = K[8*half .. 8*half+7], VGPR4-7 = K[16+8*half .. 16+8*half+7].
__device__ inline v16bf frag_b(const __bf16* base, int half) {
    union { v16bf v; v8bf h[2]; } u;
    u.h[0] = *(const v8bf*)(base + 8 * half);
    u.h[1] = *(const v8bf*)(base + 16 + 8 * half);
    return u.v;
}

// ---------------------------------------------------------------------------
// Kernel 1/4: tiled GEMM  C[M,N] = A[M,K] @ B[K,N] + bias, fp32 in/out,
// bf16 WMMA compute. BM=BN=128, BK=32, 256 threads = 8 waves,
// each wave owns a 32x64 sub-tile (2x4 WMMA accumulators).
// Register-staged double buffer: one barrier per k-step, global loads for
// tile i+1 overlap the WMMAs of tile i.
// ---------------------------------------------------------------------------
#define GBM 128
#define GBN 128
#define GBK 32
#define GPAD 40   // padded LDS row stride (elements) -> conflict-free b128 reads

__global__ void __launch_bounds__(256)
gemm_bf16_wmma(const float* __restrict__ A, const float* __restrict__ Bm,
               const float* __restrict__ bias, float* __restrict__ C,
               int M, int N, int K)
{
    __shared__ __bf16 lA[2][GBM * GPAD];   // [m][k]
    __shared__ __bf16 lB[2][GBN * GPAD];   // [n][k]  (B transposed into LDS)

    const int tid  = threadIdx.x;
    const int wave = tid >> 5;
    const int lane = tid & 31;
    const int half = lane >> 4;
    const int lrow = lane & 15;

    const int bm = blockIdx.y * GBM;
    const int bn = blockIdx.x * GBN;
    const int wm = (wave & 3) * 32;   // wave M offset within tile
    const int wn = (wave >> 2) * 64;  // wave N offset within tile

    // staging thread mapping
    const int tm = tid >> 3;          // 0..31  (A rows, +32*i)
    const int tk = (tid & 7) * 4;     // 0,4,...,28 (A k columns)
    const int kk = (tid >> 5) * 4;    // 0,4,...,28 (B k rows)
    const int tn = tid & 31;          // B n columns, +32*j

    float4 ra[4];
    float  rb[4][4];

    auto load_tiles = [&](int k0) {
#pragma unroll
        for (int i = 0; i < 4; ++i)
            ra[i] = *(const float4*)(A + (size_t)(bm + tm + 32 * i) * K + k0 + tk);
#pragma unroll
        for (int i = 0; i < 4; ++i) {
            const float* src = Bm + (size_t)(k0 + kk + i) * N + bn + tn;
#pragma unroll
            for (int j = 0; j < 4; ++j) rb[i][j] = src[32 * j];
        }
    };
    auto store_tiles = [&](int p) {
#pragma unroll
        for (int i = 0; i < 4; ++i) {
            __bf16* dst = &lA[p][(tm + 32 * i) * GPAD + tk];
            dst[0] = (__bf16)ra[i].x; dst[1] = (__bf16)ra[i].y;
            dst[2] = (__bf16)ra[i].z; dst[3] = (__bf16)ra[i].w;
        }
#pragma unroll
        for (int i = 0; i < 4; ++i)
#pragma unroll
            for (int j = 0; j < 4; ++j)
                lB[p][(tn + 32 * j) * GPAD + kk + i] = (__bf16)rb[i][j];
    };

    v8f acc[2][4];
#pragma unroll
    for (int i = 0; i < 2; ++i)
#pragma unroll
        for (int j = 0; j < 4; ++j) acc[i][j] = zero8();

    // prologue: stage tile 0
    load_tiles(0);
    store_tiles(0);
    __syncthreads();

    int p = 0;
    for (int k0 = 0; k0 < K; k0 += GBK) {
        const bool more = (k0 + GBK) < K;
        if (more) load_tiles(k0 + GBK);   // global loads overlap WMMAs below

        v16bf af[2], bf[4];
#pragma unroll
        for (int i = 0; i < 2; ++i)
            af[i] = frag_a(&lA[p][(wm + i * 16 + lrow) * GPAD], half);
#pragma unroll
        for (int j = 0; j < 4; ++j)
            bf[j] = frag_b(&lB[p][(wn + j * 16 + lrow) * GPAD], half);
#pragma unroll
        for (int i = 0; i < 2; ++i)
#pragma unroll
            for (int j = 0; j < 4; ++j)
                acc[i][j] = __builtin_amdgcn_wmma_f32_16x16x32_bf16(
                    false, af[i], false, bf[j], (short)0, acc[i][j], false, false);

        if (more) store_tiles(p ^ 1);     // waits the loads, fills other buffer
        __syncthreads();
        p ^= 1;
    }

    // epilogue: C layout -> element (m = 8*half + r, n = lrow) per 16x16 tile
#pragma unroll
    for (int i = 0; i < 2; ++i) {
#pragma unroll
        for (int j = 0; j < 4; ++j) {
            const int gn = bn + wn + j * 16 + lrow;
            const float bv = bias[gn];
#pragma unroll
            for (int r = 0; r < 8; ++r) {
                const int gm = bm + wm + i * 16 + 8 * half + r;
                C[(size_t)gm * N + gn] = acc[i][j][r] + bv;
            }
        }
    }
}

// ---------------------------------------------------------------------------
// Kernel 2: RoPE + pack fp32 [b,t,h,d] -> bf16 [b,h,t,d]
// ---------------------------------------------------------------------------
__global__ void __launch_bounds__(256)
rope_pack(const float* __restrict__ in, __bf16* __restrict__ outp, int H)
{
    const size_t total = (size_t)BB * TT * H * 64;
    for (size_t id = (size_t)blockIdx.x * blockDim.x + threadIdx.x;
         id < total; id += (size_t)gridDim.x * blockDim.x) {
        const int d = (int)(id & 63);
        size_t rest = id >> 6;
        const int h = (int)(rest % H);   rest /= H;
        const int t = (int)(rest % TT);
        const int b = (int)(rest / TT);

        const float* src = in + ((size_t)(b * TT + t) * H + h) * HD;
        const float x1 = src[d];
        const float x2 = src[d + 64];
        const float freq = __powf(1.0e6f, -(float)d * (1.0f / 64.0f));
        const float ang  = (float)t * freq;
        float s, c;
        sincosf(ang, &s, &c);
        __bf16* dst = outp + ((size_t)(b * H + h) * TT + t) * HD;
        dst[d]      = (__bf16)(x1 * c - x2 * s);
        dst[d + 64] = (__bf16)(x2 * c + x1 * s);
    }
}

// Kernel 2b: pack V fp32 [b,t,kv,d] -> bf16 transposed [b,kv,d,t]
__global__ void __launch_bounds__(256)
pack_vt(const float* __restrict__ in, __bf16* __restrict__ outp)
{
    const size_t total = (size_t)BB * NKV * HD * TT;
    for (size_t id = (size_t)blockIdx.x * blockDim.x + threadIdx.x;
         id < total; id += (size_t)gridDim.x * blockDim.x) {
        const int t = (int)(id % TT);
        size_t rest = id / TT;
        const int d  = (int)(rest % HD);  rest /= HD;
        const int kv = (int)(rest % NKV);
        const int b  = (int)(rest / NKV);
        outp[id] = (__bf16)in[((size_t)(b * TT + t) * NKV + kv) * HD + d];
    }
}

// ---------------------------------------------------------------------------
// Kernel 3: causal flash attention. Block = (b, h, 128-row q-tile),
// 8 waves x 16 q-rows each. Key tiles of 64. bf16 WMMA, fp32 online softmax.
// K/V tiles staged to LDS with async-to-LDS loads when available.
// ---------------------------------------------------------------------------
__global__ void __launch_bounds__(256)
flash_attn(const __bf16* __restrict__ Qh,  // [b,h,t,d]
           const __bf16* __restrict__ Kh,  // [b,kv,t,d]
           const __bf16* __restrict__ Vt,  // [b,kv,d,t]
           float* __restrict__ attn)       // [b,t,h*d]
{
    const int qt  = blockIdx.x;      // 0..15
    const int h   = blockIdx.y;      // 0..15
    const int b   = blockIdx.z;      // 0..1
    const int kvh = h / NREP;

    const int tid  = threadIdx.x;
    const int wave = tid >> 5;
    const int lane = tid & 31;
    const int half = lane >> 4;
    const int lrow = lane & 15;

    __shared__ __bf16 lK[64 * 136];      // [key][d], padded
    __shared__ __bf16 lV[128 * 72];      // [d][key], padded
    __shared__ __bf16 lP[8][16][64];     // per-wave probs scratch

    const int q0 = qt * 128 + wave * 16;
    const __bf16* Qbase = Qh + ((size_t)(b * NH + h) * TT + q0) * HD;
    const __bf16* Kbase = Kh + (size_t)(b * NKV + kvh) * TT * HD;
    const __bf16* Vbase = Vt + (size_t)(b * NKV + kvh) * HD * TT;

    // Q fragments: 4 chunks of K-dim (d = c*32 .. c*32+31)
    v16bf qf[4];
#pragma unroll
    for (int c = 0; c < 4; ++c)
        qf[c] = frag_a(Qbase + (size_t)lrow * HD + c * 32, half);

    v8f o[8];
#pragma unroll
    for (int j = 0; j < 8; ++j) o[j] = zero8();
    float mr[8], ls[8];
#pragma unroll
    for (int r = 0; r < 8; ++r) { mr[r] = -1.0e30f; ls[r] = 0.0f; }

    const int nkt = 2 * (qt + 1);
    for (int kt = 0; kt < nkt; ++kt) {
        const int kn0 = kt * 64;
#ifdef USE_ASYNC_LDS
        // ---- async stage: K tile [64 keys x 128 d], V^T tile [128 d x 64 keys]
        for (int cid = tid; cid < 1024; cid += 256) {
            const int key = cid >> 4, dc = cid & 15;
            __builtin_amdgcn_global_load_async_to_lds_b128(
                (as1_i32x4*)(Kbase + (size_t)(kn0 + key) * HD + dc * 8),
                (as3_i32x4*)&lK[key * 136 + dc * 8], 0, 0);
        }
        for (int cid = tid; cid < 1024; cid += 256) {
            const int d = cid >> 3, kc = cid & 7;
            __builtin_amdgcn_global_load_async_to_lds_b128(
                (as1_i32x4*)(Vbase + (size_t)d * TT + kn0 + kc * 8),
                (as3_i32x4*)&lV[d * 72 + kc * 8], 0, 0);
        }
        __builtin_amdgcn_s_wait_asynccnt(0);   // this wave's async copies done
#else
        for (int cid = tid; cid < 1024; cid += 256) {
            const int key = cid >> 4, dc = cid & 15;
            *(v8bf*)&lK[key * 136 + dc * 8] =
                *(const v8bf*)(Kbase + (size_t)(kn0 + key) * HD + dc * 8);
        }
        for (int cid = tid; cid < 1024; cid += 256) {
            const int d = cid >> 3, kc = cid & 7;
            *(v8bf*)&lV[d * 72 + kc * 8] =
                *(const v8bf*)(Vbase + (size_t)d * TT + kn0 + kc * 8);
        }
#endif
        __syncthreads();

        const bool active = (kn0 <= q0 + 15);   // causal: skip tiles fully above diag
        if (active) {
            // ---- S = Q K^T  (4 n-tiles of 16 keys, K-dim = 128 in 4 chunks) ----
            v8f s[4];
#pragma unroll
            for (int j = 0; j < 4; ++j) s[j] = zero8();
#pragma unroll
            for (int j = 0; j < 4; ++j)
#pragma unroll
                for (int c = 0; c < 4; ++c) {
                    const v16bf bk = frag_b(&lK[(j * 16 + lrow) * 136 + c * 32], half);
                    s[j] = __builtin_amdgcn_wmma_f32_16x16x32_bf16(
                        false, qf[c], false, bk, (short)0, s[j], false, false);
                }
            // ---- scale + causal mask ----
            const float scl = 0.08838834764831845f;  // 1/sqrt(128)
#pragma unroll
            for (int j = 0; j < 4; ++j) {
                const int kg = kn0 + j * 16 + lrow;   // key index for this lane
#pragma unroll
                for (int r = 0; r < 8; ++r) {
                    const int qg = q0 + 8 * half + r;
                    s[j][r] = (kg > qg) ? -1.0e30f : s[j][r] * scl;
                }
            }
            // ---- online softmax (row = 8*half + r; 16-lane butterfly reduce) ----
#pragma unroll
            for (int r = 0; r < 8; ++r) {
                float rmax = -1.0e30f;
#pragma unroll
                for (int j = 0; j < 4; ++j) rmax = fmaxf(rmax, s[j][r]);
#pragma unroll
                for (int mk = 1; mk < 16; mk <<= 1)
                    rmax = fmaxf(rmax, __shfl_xor(rmax, mk, 32));
                const float mnew = fmaxf(mr[r], rmax);
                const float corr = __expf(mr[r] - mnew);
                mr[r] = mnew;
                float rsum = 0.0f;
#pragma unroll
                for (int j = 0; j < 4; ++j) {
                    const float p = __expf(s[j][r] - mnew);
                    s[j][r] = p;
                    rsum += p;
                }
#pragma unroll
                for (int mk = 1; mk < 16; mk <<= 1)
                    rsum += __shfl_xor(rsum, mk, 32);
                ls[r] = ls[r] * corr + rsum;
#pragma unroll
                for (int jt = 0; jt < 8; ++jt) o[jt][r] *= corr;
            }
            // ---- write P (bf16) to per-wave LDS scratch in row-major ----
#pragma unroll
            for (int j = 0; j < 4; ++j)
#pragma unroll
                for (int r = 0; r < 8; ++r)
                    lP[wave][8 * half + r][j * 16 + lrow] = (__bf16)s[j][r];
        }
        __syncthreads();   // cross-lane P visibility (and uniform across block)

        if (active) {
            // ---- O += P @ V : A = P (16 x 64 in 2 K-chunks), B = V^T tile ----
#pragma unroll
            for (int kc = 0; kc < 2; ++kc) {
                const v16bf ap = frag_a(&lP[wave][lrow][kc * 32], half);
#pragma unroll
                for (int jt = 0; jt < 8; ++jt) {
                    const v16bf bv = frag_b(&lV[(jt * 16 + lrow) * 72 + kc * 32], half);
                    o[jt] = __builtin_amdgcn_wmma_f32_16x16x32_bf16(
                        false, ap, false, bv, (short)0, o[jt], false, false);
                }
            }
        }
        __syncthreads();   // protect lK/lV before next stage
    }

    // ---- epilogue: normalize and store attn[b, t, h*128 + d] (fp32) ----
#pragma unroll
    for (int r = 0; r < 8; ++r) {
        const float inv = 1.0f / ls[r];
        const int gm = q0 + 8 * half + r;
#pragma unroll
        for (int jt = 0; jt < 8; ++jt) {
            const int gn = h * HD + jt * 16 + lrow;
            attn[((size_t)(b * TT + gm)) * NE + gn] = o[jt][r] * inv;
        }
    }
}

// ---------------------------------------------------------------------------
// Host launcher
// ---------------------------------------------------------------------------
extern "C" void kernel_launch(void* const* d_in, const int* in_sizes, int n_in,
                              void* d_out, int out_size, void* d_ws, size_t ws_size,
                              hipStream_t stream)
{
    (void)in_sizes; (void)n_in; (void)out_size; (void)ws_size;
    const float* x  = (const float*)d_in[0];
    const float* Wq = (const float*)d_in[1];
    const float* bq = (const float*)d_in[2];
    const float* Wk = (const float*)d_in[3];
    const float* bk = (const float*)d_in[4];
    const float* Wv = (const float*)d_in[5];
    const float* bv = (const float*)d_in[6];
    const float* Wo = (const float*)d_in[7];
    const float* bo = (const float*)d_in[8];
    float* out = (float*)d_out;

    const size_t MT = (size_t)BB * TT;       // 4096 rows
    float*  Qf  = (float*)d_ws;              // [b,t,16*128] fp32
    float*  Kf  = Qf + MT * NE;              // [b,t,4*128]  fp32
    float*  Vf  = Kf + MT * (NKV * HD);      // [b,t,4*128]  fp32
    __bf16* Qh  = (__bf16*)(Vf + MT * (NKV * HD));  // [b,h,t,d]
    __bf16* Khd = Qh + MT * NE;                     // [b,kv,t,d]
    __bf16* Vtd = Khd + MT * (NKV * HD);            // [b,kv,d,t]
    float*  attn = Qf;   // reuse Q fp32 region for attention output

    dim3 blk(256);
    // QKV projections (bf16 WMMA GEMMs)
    gemm_bf16_wmma<<<dim3(NE / GBN, MT / GBM), blk, 0, stream>>>(x, Wq, bq, Qf, (int)MT, NE, NE);
    gemm_bf16_wmma<<<dim3((NKV * HD) / GBN, MT / GBM), blk, 0, stream>>>(x, Wk, bk, Kf, (int)MT, NKV * HD, NE);
    gemm_bf16_wmma<<<dim3((NKV * HD) / GBN, MT / GBM), blk, 0, stream>>>(x, Wv, bv, Vf, (int)MT, NKV * HD, NE);
    // RoPE + pack to WMMA-friendly bf16 layouts
    rope_pack<<<dim3(16384), blk, 0, stream>>>(Qf, Qh, NH);
    rope_pack<<<dim3(4096),  blk, 0, stream>>>(Kf, Khd, NKV);
    pack_vt<<<dim3(8192),    blk, 0, stream>>>(Vf, Vtd);
    // Causal flash attention
    flash_attn<<<dim3(TT / 128, NH, BB), blk, 0, stream>>>(Qh, Khd, Vtd, attn);
    // Output projection
    gemm_bf16_wmma<<<dim3(NE / GBN, MT / GBM), blk, 0, stream>>>(attn, Wo, bo, out, (int)MT, NE, NE);
}
